// HGCN4MeSH_27264452395400
// MI455X (gfx1250) — matrix operand order repaired and checked by reading, not compiled, not run on hardware
//
#include <hip/hip_runtime.h>

// ---------------------------------------------------------------------------
// HGCN4MeSH forward for MI455X (gfx1250, wave32, WMMA bf16 w/ f32 accumulate)
//
// ~100 GFLOP total, dominated by doc-label attention GEMMs; label_feature bf16
// (23MB padded) is L2-resident (192MB L2) -> matrix-unit bound -> WMMA bf16.
// All WMMA operands are zero-padded to K%32==0 so fragment loads are
// unconditional 16B/32B vector loads (no per-element EXEC branching).
// feat [B,28000,400] (179MB) is never materialized: attention + softmax +
// weighted sum + FC1 + FC2 fused per (batch, 16-label tile) in LDS.
// GRU recurrence is latency bound -> persistent 256-thread scan per direction.
// ---------------------------------------------------------------------------

typedef __bf16 bf16;
typedef __attribute__((ext_vector_type(16))) __bf16 v16bf;
typedef __attribute__((ext_vector_type(8)))  __bf16 v8bf;
typedef __attribute__((ext_vector_type(8)))  float  v8f;

#define NLAB    28000
#define BATCH   4
#define T_ABS   384
#define T_TIT   64
#define NEDGE   224000

// ---------------- WMMA helpers (CDNA5 16x16x32 bf16, f32 acc) --------------

__device__ inline v8f wmma_bf16(v16bf a, v16bf b, v8f c) {
  return __builtin_amdgcn_wmma_f32_16x16x32_bf16(false, a, false, b, (short)0, c,
                                                 false, false);
}

// A fragment, padded row-major source (ld % 16 == 0 elements, no guards).
// ISA 16-bit A 16x32: lanes 0-15 row=lane, K {k0..k0+7, k0+16..k0+23};
// lanes 16-31 same rows, K {k0+8..k0+15, k0+24..k0+31}.
__device__ inline v16bf ld_a_pad(const bf16* __restrict__ A, int ld, int row,
                                 int k0, int lane) {
  const bf16* p = A + (long)row * ld + k0 + ((lane >> 4) & 1) * 8;
  v8bf lo = *(const v8bf*)p;
  v8bf hi = *(const v8bf*)(p + 16);
  v16bf v;
#pragma unroll
  for (int e = 0; e < 8; ++e) { v[e] = lo[e]; v[e + 8] = hi[e]; }
  return v;
}

// A fragment from f32 LDS tile (converted on the fly), no guards.
__device__ inline v16bf ld_a_lds(const float* __restrict__ P, int ld, int row,
                                 int k0, int lane) {
  const float* p = P + row * ld + k0 + ((lane >> 4) & 1) * 8;
  v16bf v;
#pragma unroll
  for (int e = 0; e < 8; ++e) { v[e] = (bf16)p[e]; v[e + 8] = (bf16)p[16 + e]; }
  return v;
}

// B fragment where B[k][n] = W[n][k], W padded row-major [Npad, ld] (ld%16==0).
// ISA 16-bit B 32x16: lane = column n, lanes 0-15 K {k0..k0+7, k0+16..k0+23};
// lanes 16-31 K {k0+8..k0+15, k0+24..k0+31} -> two 16-byte loads per lane.
__device__ inline v16bf ld_b_pad(const bf16* __restrict__ W, int ld, int n0,
                                 int k0, int lane) {
  const bf16* p = W + (long)(n0 + (lane & 15)) * ld + k0 + ((lane >> 4) & 1) * 8;
  v8bf lo = *(const v8bf*)p;
  v8bf hi = *(const v8bf*)(p + 16);
  v16bf v;
#pragma unroll
  for (int e = 0; e < 8; ++e) { v[e] = lo[e]; v[e + 8] = hi[e]; }
  return v;
}

// ---------------- elementwise / utility kernels ----------------------------

__global__ void k_zero_f32(float* __restrict__ p, int n) {
  int i = blockIdx.x * blockDim.x + threadIdx.x;
  if (i < n) p[i] = 0.0f;
}

// padded f32 -> bf16: out[r][c] over [Rout x Cout], zero outside [R x C]
__global__ void k_pad_bf16(const float* __restrict__ in, bf16* __restrict__ out,
                           int R, int C, int Cout, int total) {
  int i = blockIdx.x * blockDim.x + threadIdx.x;
  if (i >= total) return;
  int r = i / Cout, c = i - r * Cout;
  out[i] = (r < R && c < C) ? (bf16)in[(long)r * C + c] : (bf16)0.0f;
}

// GCN message passing: agg[dst] += feat[src]
__global__ void k_scatter_add(const float* __restrict__ feat,
                              const int* __restrict__ src,
                              const int* __restrict__ dst,
                              float* __restrict__ agg, int total, int D) {
  int i = blockIdx.x * blockDim.x + threadIdx.x;
  if (i >= total) return;
  int e = i / D, d = i - e * D;
  atomicAdd(&agg[(long)dst[e] * D + d], feat[(long)src[e] * D + d]);
}

__global__ void k_embed(const int* __restrict__ tok, const float* __restrict__ emb,
                        float* __restrict__ x, int total, int D) {
  int i = blockIdx.x * blockDim.x + threadIdx.x;
  if (i >= total) return;
  int t = i / D, d = i - t * D;
  x[i] = emb[(long)tok[t] * D + d];
}

// label_feature = concat(lab, g_node) -> bf16 [NLAB, 416], cols 400..415 = 0
__global__ void k_pack_label(const float* __restrict__ lab,
                             const float* __restrict__ gn,
                             bf16* __restrict__ L, int total) {
  int i = blockIdx.x * blockDim.x + threadIdx.x;
  if (i >= total) return;
  int n = i / 416, c = i - n * 416;
  float v = (c < 200) ? lab[(long)n * 200 + c]
          : (c < 400) ? gn[(long)n * 200 + (c - 200)] : 0.0f;
  L[i] = (bf16)v;
}

// Xtr[b][d][t] = in[b][t][d]  (bf16 transposed copy for the feat GEMM B side)
__global__ void k_transpose_bf16(const float* __restrict__ in, bf16* __restrict__ out,
                                 int T, int total) {
  int i = blockIdx.x * blockDim.x + threadIdx.x;
  if (i >= total) return;
  int b = i / (400 * T), rem = i - b * 400 * T;
  int d = rem / T, t = rem - d * T;
  out[i] = (bf16)in[((long)b * T + t) * 400 + d];
}

// out[m][n] = bias[n] + A[m]·W[n]   (f32, float4 inner, for GRU gate GEMM)
__global__ void k_linear_f32(const float* __restrict__ A,
                             const float* __restrict__ W,
                             const float* __restrict__ bias,
                             float* __restrict__ out, int M, int N, int K) {
  int i = blockIdx.x * blockDim.x + threadIdx.x;
  if (i >= M * N) return;
  int m = i / N, n = i - m * N;
  const float4* a = (const float4*)(A + (long)m * K);
  const float4* w = (const float4*)(W + (long)n * K);
  float acc = bias[n];
  int K4 = K >> 2;
  for (int k = 0; k < K4; ++k) {
    float4 av = a[k], wv = w[k];
    acc += av.x * wv.x + av.y * wv.y + av.z * wv.z + av.w * wv.w;
  }
  out[i] = acc;
}

// ---------------- WMMA GEMM: C = act(A @ W^T + bias) (GCN layers) ----------
// A: [M, Kpad] bf16 padded; W: [Npad, Kpad] bf16 padded; C: [M, N] f32.

__global__ __launch_bounds__(32) void k_gemm_wt_bf16(
    const bf16* __restrict__ A, const bf16* __restrict__ W,
    const float* __restrict__ bias, float* __restrict__ C,
    int M, int N, int Kpad, int act) {
  int m0 = blockIdx.x * 16, n0 = blockIdx.y * 16;
  int lane = threadIdx.x;
  v8f acc;
#pragma unroll
  for (int r = 0; r < 8; ++r) acc[r] = 0.0f;
  for (int k0 = 0; k0 < Kpad; k0 += 32) {
    v16bf a = ld_a_pad(A, Kpad, m0 + (lane & 15), k0, lane);
    v16bf b = ld_b_pad(W, Kpad, n0, k0, lane);
    acc = wmma_bf16(a, b, acc);
  }
  int n = n0 + (lane & 15);
  int mb = m0 + ((lane >> 4) & 1) * 8;
  if (n < N) {
    float bv = bias[n];
#pragma unroll
    for (int r = 0; r < 8; ++r) {
      int m = mb + r;
      if (m < M) {
        float v = acc[r] + bv;
        if (act == 1) v = v > 0.0f ? v : 0.0f;
        C[(long)m * N + n] = v;
      }
    }
  }
}

// ---------------- persistent GRU scan (one block per direction) ------------

__global__ void k_gru_scan(const float* __restrict__ gx,   // [B,T,600]
                           const float* __restrict__ whh,  // [600,200]
                           const float* __restrict__ bhh,  // [600]
                           const int* __restrict__ len,    // [B]
                           float* __restrict__ out,        // [B,T,400]
                           int T, int rev, int colOff) {
  const int H = 200, B = BATCH, G = 600;
  __shared__ __align__(16) float h[BATCH * 200];
  __shared__ float gh[BATCH * 600];
  __shared__ int tix[BATCH];
  __shared__ int msk[BATCH];
  int tid = threadIdx.x;
  for (int i = tid; i < B * H; i += blockDim.x) h[i] = 0.0f;
  __syncthreads();
  for (int s = 0; s < T; ++s) {
    if (tid < B) {
      int L = len[tid];
      int m = (s < L) ? 1 : 0;
      msk[tid] = m;
      tix[tid] = rev ? (m ? (L - 1 - s) : s) : s;
    }
    __syncthreads();
    for (int idx = tid; idx < B * G; idx += blockDim.x) {
      int b = idx / G, j = idx - b * G;
      const float4* w4 = (const float4*)(whh + (long)j * H);
      const float4* h4 = (const float4*)(h + b * H);
      float acc = bhh[j];
      for (int k = 0; k < H / 4; ++k) {
        float4 wv = w4[k], hv = h4[k];
        acc += hv.x * wv.x + hv.y * wv.y + hv.z * wv.z + hv.w * wv.w;
      }
      gh[idx] = acc;
    }
    __syncthreads();
    for (int idx = tid; idx < B * H; idx += blockDim.x) {
      int b = idx / H, i = idx - b * H;
      const float* g = gx + ((long)b * T + tix[b]) * G;
      float r = 1.0f / (1.0f + __expf(-(g[i] + gh[b * G + i])));
      float z = 1.0f / (1.0f + __expf(-(g[H + i] + gh[b * G + H + i])));
      float nn = tanhf(g[2 * H + i] + r * gh[b * G + 2 * H + i]);
      float hold = h[idx];
      float hn = (1.0f - z) * nn + z * hold;
      float o;
      if (msk[b]) { o = hn; } else { o = 0.0f; hn = hold; }
      h[idx] = hn;
      out[((long)b * T + tix[b]) * 400 + colOff + i] = o;
    }
    __syncthreads();
  }
}

// ---------------- fused attention (both branches) + FC1 + FC2 --------------
// One wave per (batch b, 16-label tile); LDS = 64000 B.

__global__ __launch_bounds__(32) void k_attn_out(
    const bf16* __restrict__ L,     // [NLAB, 416]
    const bf16* __restrict__ Xt,    // [B, 64, 416]
    const bf16* __restrict__ XtrT,  // [B, 400, 64]
    const bf16* __restrict__ Xa,    // [B, 384, 416]
    const bf16* __restrict__ XtrA,  // [B, 400, 384]
    const bf16* __restrict__ fc1w,  // [208, 416]
    const float* __restrict__ fc1b, // [200]
    const float* __restrict__ fc2w, // [200]
    const float* __restrict__ fc2b, // [1]
    float* __restrict__ out) {      // [B, NLAB]
  __shared__ float S[16 * 384];             // logits -> attention weights
  __shared__ __align__(16) float F[16 * 416];  // feat tile (cols 400.. stay 0)
  __shared__ float H1[16 * 200];            // fc1 activations
  int n0 = blockIdx.x * 16;
  int b = blockIdx.y;
  int lane = threadIdx.x;

  for (int i = lane; i < 16 * 416; i += 32) F[i] = 0.0f;
  __syncthreads();

  for (int br = 0; br < 2; ++br) {
    const bf16* Xb = (br ? Xa : Xt) + (long)b * (br ? T_ABS : T_TIT) * 416;
    const bf16* Xtrb = (br ? XtrA : XtrT) + (long)b * 400 * (br ? T_ABS : T_TIT);
    const int T = br ? T_ABS : T_TIT;

    // logits S[m][t] = L[n0+m] . x[b,t]   (M=16 labels, N=t, K=416 padded)
    for (int tt = 0; tt < T; tt += 16) {
      v8f acc;
#pragma unroll
      for (int r = 0; r < 8; ++r) acc[r] = 0.0f;
      for (int k0 = 0; k0 < 416; k0 += 32) {
        v16bf a = ld_a_pad(L, 416, n0 + (lane & 15), k0, lane);
        v16bf bb = ld_b_pad(Xb, 416, tt, k0, lane);
        acc = wmma_bf16(a, bb, acc);
      }
      int t = tt + (lane & 15);
      int mb = ((lane >> 4) & 1) * 8;
#pragma unroll
      for (int r = 0; r < 8; ++r) S[(mb + r) * 384 + t] = acc[r];
    }
    __syncthreads();

    // softmax over t per label row (padded t -> logit 0, included, as in ref)
    if (lane < 16) {
      float* row = S + lane * 384;
      float mx = -3.4e38f;
      for (int t = 0; t < T; ++t) mx = fmaxf(mx, row[t]);
      float sum = 0.0f;
      for (int t = 0; t < T; ++t) { float e = __expf(row[t] - mx); row[t] = e; sum += e; }
      float inv = 1.0f / sum;
      for (int t = 0; t < T; ++t) row[t] *= inv;
    }
    __syncthreads();

    // feat[m][d] += sum_t atten[m][t] * x[b,t,d]   (M=16, N=400, K=T)
    for (int d0 = 0; d0 < 400; d0 += 16) {
      v8f acc;
#pragma unroll
      for (int r = 0; r < 8; ++r) acc[r] = 0.0f;
      for (int t0 = 0; t0 < T; t0 += 32) {
        v16bf a = ld_a_lds(S, 384, lane & 15, t0, lane);
        v16bf bb = ld_b_pad(Xtrb, T, d0, t0, lane);
        acc = wmma_bf16(a, bb, acc);
      }
      int d = d0 + (lane & 15);
      int mb = ((lane >> 4) & 1) * 8;
#pragma unroll
      for (int r = 0; r < 8; ++r) F[(mb + r) * 416 + d] += acc[r];
    }
    __syncthreads();
  }

  // FC1: H1 = leaky_relu(F @ fc1w^T + fc1b)   (M=16, N=200, K=416 padded)
  for (int c0 = 0; c0 < 208; c0 += 16) {
    v8f acc;
#pragma unroll
    for (int r = 0; r < 8; ++r) acc[r] = 0.0f;
    for (int k0 = 0; k0 < 416; k0 += 32) {
      v16bf a = ld_a_lds(F, 416, lane & 15, k0, lane);
      v16bf bb = ld_b_pad(fc1w, 416, c0, k0, lane);
      acc = wmma_bf16(a, bb, acc);
    }
    int c = c0 + (lane & 15);
    int mb = ((lane >> 4) & 1) * 8;
    if (c < 200) {
      float bv = fc1b[c];
#pragma unroll
      for (int r = 0; r < 8; ++r) {
        float v = acc[r] + bv;
        H1[(mb + r) * 200 + c] = v > 0.0f ? v : 0.2f * v;
      }
    }
  }
  __syncthreads();

  // FC2 + leaky_relu -> y[b, n0+m]
  if (lane < 16) {
    const float* hr = H1 + lane * 200;
    float acc = fc2b[0];
    for (int n = 0; n < 200; ++n) acc += hr[n] * fc2w[n];
    float y = acc > 0.0f ? acc : 0.2f * acc;
    out[(long)b * NLAB + n0 + lane] = y;
  }
}

// ---------------------------------------------------------------------------

extern "C" void kernel_launch(void* const* d_in, const int* in_sizes, int n_in,
                              void* d_out, int out_size, void* d_ws, size_t ws_size,
                              hipStream_t stream) {
  const int* input_abstract = (const int*)d_in[0];
  const int* input_title    = (const int*)d_in[1];
  const int* ab_length      = (const int*)d_in[2];
  const int* title_length   = (const int*)d_in[3];
  const int* edge_src       = (const int*)d_in[4];
  const int* edge_dst       = (const int*)d_in[5];
  const float* g_node = (const float*)d_in[6];
  const float* emb    = (const float*)d_in[7];
  const float* wih_l0f = (const float*)d_in[8];
  const float* whh_l0f = (const float*)d_in[9];
  const float* bih_l0f = (const float*)d_in[10];
  const float* bhh_l0f = (const float*)d_in[11];
  const float* wih_l0b = (const float*)d_in[12];
  const float* whh_l0b = (const float*)d_in[13];
  const float* bih_l0b = (const float*)d_in[14];
  const float* bhh_l0b = (const float*)d_in[15];
  const float* wih_l1f = (const float*)d_in[16];
  const float* whh_l1f = (const float*)d_in[17];
  const float* bih_l1f = (const float*)d_in[18];
  const float* bhh_l1f = (const float*)d_in[19];
  const float* wih_l1b = (const float*)d_in[20];
  const float* whh_l1b = (const float*)d_in[21];
  const float* bih_l1b = (const float*)d_in[22];
  const float* bhh_l1b = (const float*)d_in[23];
  const float* gcn_w1 = (const float*)d_in[24];
  const float* gcn_b1 = (const float*)d_in[25];
  const float* gcn_w2 = (const float*)d_in[26];
  const float* gcn_b2 = (const float*)d_in[27];
  const float* fc1_w = (const float*)d_in[28];
  const float* fc1_b = (const float*)d_in[29];
  const float* fc2_w = (const float*)d_in[30];
  const float* fc2_b = (const float*)d_in[31];
  (void)in_sizes; (void)n_in; (void)out_size; (void)ws_size;

  // ---- workspace carve-out (fixed deterministic layout, ~100MB) ----
  char* base = (char*)d_ws;
  size_t off = 0;
  auto take = [&](size_t bytes) -> void* {
    void* p = base + off;
    off += (bytes + 255) & ~(size_t)255;
    return p;
  };
  float* agg   = (float*)take((size_t)NLAB * 200 * 4);  // reused as 'lab'
  bf16*  aggb  = (bf16*) take((size_t)NLAB * 224 * 2);
  float* hbuf  = (float*)take((size_t)NLAB * 200 * 4);
  bf16*  w1p   = (bf16*) take(208 * 224 * 2);
  bf16*  w2p   = (bf16*) take(208 * 224 * 2);
  bf16*  fc1wp = (bf16*) take(208 * 416 * 2);
  bf16*  Lb    = (bf16*) take((size_t)NLAB * 416 * 2);
  float* x0    = (float*)take((size_t)BATCH * T_ABS * 200 * 4);
  float* gxf   = (float*)take((size_t)BATCH * T_ABS * 600 * 4);
  float* gxb   = (float*)take((size_t)BATCH * T_ABS * 600 * 4);
  float* out0  = (float*)take((size_t)BATCH * T_ABS * 400 * 4);
  float* out1  = (float*)take((size_t)BATCH * T_ABS * 400 * 4);
  bf16*  xbT   = (bf16*) take((size_t)BATCH * T_TIT * 416 * 2);
  bf16*  xbA   = (bf16*) take((size_t)BATCH * T_ABS * 416 * 2);
  bf16*  xtrT  = (bf16*) take((size_t)BATCH * 400 * T_TIT * 2);
  bf16*  xtrA  = (bf16*) take((size_t)BATCH * 400 * T_ABS * 2);

  const int THR = 256;
  auto g1 = [&](long n) { return dim3((unsigned)((n + THR - 1) / THR)); };

  // ---- weight padding/conversion ----
  k_pad_bf16<<<g1(208 * 224), THR, 0, stream>>>(gcn_w1, w1p, 200, 200, 224, 208 * 224);
  k_pad_bf16<<<g1(208 * 224), THR, 0, stream>>>(gcn_w2, w2p, 200, 200, 224, 208 * 224);
  k_pad_bf16<<<g1(208 * 416), THR, 0, stream>>>(fc1_w, fc1wp, 200, 400, 416, 208 * 416);

  // ---- GCN layer 1 ----
  const long aggN = (long)NLAB * 200;
  k_zero_f32<<<g1(aggN), THR, 0, stream>>>(agg, (int)aggN);
  k_scatter_add<<<g1((long)NEDGE * 200), THR, 0, stream>>>(
      g_node, edge_src, edge_dst, agg, NEDGE * 200, 200);
  k_pad_bf16<<<g1((long)NLAB * 224), THR, 0, stream>>>(agg, aggb, NLAB, 200, 224, NLAB * 224);
  k_gemm_wt_bf16<<<dim3(NLAB / 16, 13), 32, 0, stream>>>(
      aggb, w1p, gcn_b1, hbuf, NLAB, 200, 224, /*relu*/ 1);

  // ---- GCN layer 2 (lab written into agg) ----
  k_zero_f32<<<g1(aggN), THR, 0, stream>>>(agg, (int)aggN);
  k_scatter_add<<<g1((long)NEDGE * 200), THR, 0, stream>>>(
      hbuf, edge_src, edge_dst, agg, NEDGE * 200, 200);
  k_pad_bf16<<<g1((long)NLAB * 224), THR, 0, stream>>>(agg, aggb, NLAB, 200, 224, NLAB * 224);
  k_gemm_wt_bf16<<<dim3(NLAB / 16, 13), 32, 0, stream>>>(
      aggb, w2p, gcn_b2, agg, NLAB, 200, 224, /*none*/ 0);
  k_pack_label<<<g1((long)NLAB * 416), THR, 0, stream>>>(agg, g_node, Lb, NLAB * 416);

  // ---- GRU branches (title T=64, abstract T=384) ----
  for (int br = 0; br < 2; ++br) {
    const int T = br ? T_ABS : T_TIT;
    const int* tok = br ? input_abstract : input_title;
    const int* len = br ? ab_length : title_length;
    bf16* xb = br ? xbA : xbT;
    bf16* xtr = br ? xtrA : xtrT;
    const int BT = BATCH * T;

    k_embed<<<g1((long)BT * 200), THR, 0, stream>>>(tok, emb, x0, BT * 200, 200);
    // layer 0
    k_linear_f32<<<g1((long)BT * 600), THR, 0, stream>>>(x0, wih_l0f, bih_l0f, gxf, BT, 600, 200);
    k_linear_f32<<<g1((long)BT * 600), THR, 0, stream>>>(x0, wih_l0b, bih_l0b, gxb, BT, 600, 200);
    k_gru_scan<<<1, 256, 0, stream>>>(gxf, whh_l0f, bhh_l0f, len, out0, T, 0, 0);
    k_gru_scan<<<1, 256, 0, stream>>>(gxb, whh_l0b, bhh_l0b, len, out0, T, 1, 200);
    // layer 1 (gx buffers reused)
    k_linear_f32<<<g1((long)BT * 600), THR, 0, stream>>>(out0, wih_l1f, bih_l1f, gxf, BT, 600, 400);
    k_linear_f32<<<g1((long)BT * 600), THR, 0, stream>>>(out0, wih_l1b, bih_l1b, gxb, BT, 600, 400);
    k_gru_scan<<<1, 256, 0, stream>>>(gxf, whh_l1f, bhh_l1f, len, out1, T, 0, 0);
    k_gru_scan<<<1, 256, 0, stream>>>(gxb, whh_l1b, bhh_l1b, len, out1, T, 1, 200);

    k_pad_bf16<<<g1((long)BT * 416), THR, 0, stream>>>(out1, xb, BT, 400, 416, BT * 416);
    k_transpose_bf16<<<g1((long)BATCH * 400 * T), THR, 0, stream>>>(out1, xtr, T, BATCH * 400 * T);
  }

  // ---- fused attention (both branches) + FC1 + FC2 -> y[B, NLAB] ----
  k_attn_out<<<dim3(NLAB / 16, BATCH), 32, 0, stream>>>(
      Lb, xbT, xtrT, xbA, xtrA, fc1wp, fc1_b, fc2_w, fc2_b, (float*)d_out);
}
